// MixtureOfExperts_77455440216219
// MI455X (gfx1250) — matrix operand order repaired and checked by reading, not compiled, not run on hardware
//
#include <hip/hip_runtime.h>
#include <math.h>

typedef __attribute__((ext_vector_type(16))) _Float16 v16h;
typedef __attribute__((ext_vector_type(8)))  float    v8f;

#define B_SEQ 4096
#define D_IN  128
#define H_DIM 256
#define G4    1024   // 4*H
#define OUTD  16
#define NE    16

#if __has_builtin(__builtin_amdgcn_global_load_async_to_lds_b128) && \
    __has_builtin(__builtin_amdgcn_s_wait_asynccnt)
#define USE_ASYNC 1
typedef int vi4 __attribute__((vector_size(16)));
typedef __attribute__((address_space(1))) vi4* g_vi4p;   // global
typedef __attribute__((address_space(3))) vi4* l_vi4p;   // LDS
#else
#define USE_ASYNC 0
#endif

// ---------------------------------------------------------------------------
// Kernel 0: zero the output accumulator (harness poisons d_out).
// ---------------------------------------------------------------------------
__global__ void zero_out_kernel(float* __restrict__ out, int n) {
  int i = blockIdx.x * blockDim.x + threadIdx.x;
  if (i < n) out[i] = 0.0f;
}

// ---------------------------------------------------------------------------
// Kernel 1: gating. One thread per row: scores -> softmax -> top-2 mask.
// ---------------------------------------------------------------------------
__global__ void gating_kernel(const float* __restrict__ x,
                              const float* __restrict__ Wg,
                              const float* __restrict__ bg,
                              float* __restrict__ gated) {
  int b = blockIdx.x * blockDim.x + threadIdx.x;
  if (b >= B_SEQ) return;
  const float* xr = x + (size_t)b * D_IN;
  float sc[NE];
  for (int e = 0; e < NE; ++e) {
    float s = bg[e];
    const float* w = Wg + e * D_IN;
    for (int d = 0; d < D_IN; ++d) s += xr[d] * w[d];
    sc[e] = s;
  }
  float m = sc[0];
  for (int e = 1; e < NE; ++e) m = fmaxf(m, sc[e]);
  float sum = 0.f;
  for (int e = 0; e < NE; ++e) { sc[e] = __expf(sc[e] - m); sum += sc[e]; }
  float inv = 1.0f / sum;
  for (int e = 0; e < NE; ++e) sc[e] *= inv;
  int i1 = 0;
  for (int e = 1; e < NE; ++e) if (sc[e] > sc[i1]) i1 = e;
  int i2 = (i1 == 0) ? 1 : 0;
  for (int e = 0; e < NE; ++e) if (e != i1 && sc[e] > sc[i2]) i2 = e;
  for (int e = 0; e < NE; ++e)
    gated[b * NE + e] = (e == i1 || e == i2) ? sc[e] : 0.0f;
}

// ---------------------------------------------------------------------------
// Kernel 2: input projection GEMM via WMMA f16.
//   xg[e][t][r] = sum_d W_ih[e][r][d] * x[t][d] + b_ih[e][r] + b_hh[e][r]
// ---------------------------------------------------------------------------
__global__ void __launch_bounds__(256)
xproj_kernel(const float* __restrict__ x,     // [4096,128]
             const float* __restrict__ W_ih,  // [16,1024,128]
             const float* __restrict__ b_ih,  // [16,1024]
             const float* __restrict__ b_hh,  // [16,1024]
             float* __restrict__ xg)          // [16,4096,1024]
{
  int wave = threadIdx.x >> 5;
  int lane = threadIdx.x & 31;
  int hf   = lane >> 4;
  int l15  = lane & 15;

  int tile = blockIdx.x * (blockDim.x >> 5) + wave;   // 16*256*64 tiles
  int e   = tile >> 14;
  int rem = tile & 16383;
  int tt  = rem >> 6;
  int rt  = rem & 63;

  const float* Wrow = W_ih + ((size_t)e * G4 + (size_t)(rt * 16 + l15)) * D_IN;
  const float* Xrow = x + (size_t)(tt * 16 + l15) * D_IN;

  v8f acc = {};
  #pragma unroll
  for (int kt = 0; kt < D_IN / 32; ++kt) {
    v16h a, b;
    #pragma unroll
    for (int v = 0; v < 8; ++v) {
      int ka = kt * 32 + ((v >> 2) << 4) + (hf << 3) + ((v & 3) << 1);
      a[2 * v]     = (_Float16)Wrow[ka];
      a[2 * v + 1] = (_Float16)Wrow[ka + 1];
      int kb = kt * 32 + ((v >> 2) << 4) + (hf << 3) + ((v & 3) << 1);
      b[2 * v]     = (_Float16)Xrow[kb];
      b[2 * v + 1] = (_Float16)Xrow[kb + 1];
    }
    acc = __builtin_amdgcn_wmma_f32_16x16x32_f16(false, a, false, b,
                                                 (short)0, acc, false, false);
  }
  int t = tt * 16 + l15;
  int rbase = rt * 16 + hf * 8;
  float* dst = xg + ((size_t)e * B_SEQ + t) * G4 + rbase;
  const float* bi = b_ih + e * G4 + rbase;
  const float* bh = b_hh + e * G4 + rbase;
  #pragma unroll
  for (int r = 0; r < 8; ++r) dst[r] = acc[r] + bi[r] + bh[r];
}

// ---------------------------------------------------------------------------
// Kernel 3: persistent LSTM scan. 16 blocks (one per expert), 512 thr (16 waves).
// W_hh held in VGPRs as f16 A-fragments. Per step:
//   - waves 8..15 async-stage xg[t+1] into an LDS double buffer
//     (GLOBAL_LOAD_ASYNC_TO_LDS_B128, ASYNCcnt pipeline depth 1)
//   - GEMV via WMMA, h broadcast into all 16 B columns (B built once per Ktile)
//   - gate math + c/h update + fused W_lin projection + gated atomicAdd
// ---------------------------------------------------------------------------
__global__ void __launch_bounds__(512, 1)
lstm_scan_kernel(const float* __restrict__ W_hh,   // [16,1024,256]
                 const float* __restrict__ W_lin,  // [16,16,256]
                 const float* __restrict__ b_lin,  // [16,16]
                 const float* __restrict__ xg,     // [16,4096,1024]
                 const float* __restrict__ gated,  // [4096,16]
                 float* __restrict__ out)          // [4096,16]
{
  __shared__ float    gpre[G4];
  __shared__ float    hbuf[H_DIM];
  __shared__ _Float16 hh16[H_DIM];
  __shared__ float    wlin_s[OUTD * H_DIM];
  __shared__ float    red[256];
#if USE_ASYNC
  __shared__ float    xbuf[2][G4];        // async-staged xg[t], xg[t+1]
#endif

  const int e    = blockIdx.x;
  const int wave = threadIdx.x >> 5;
  const int lane = threadIdx.x & 31;
  const int hf   = lane >> 4;
  const int l15  = lane & 15;

  const float* xg_e = xg + (size_t)e * B_SEQ * G4;

#if USE_ASYNC
  // Stage xg[0] into xbuf[0] (waves 8..15; 16B per lane).
  if (threadIdx.x >= H_DIM) {
    int j = threadIdx.x - H_DIM;          // 0..255
    __builtin_amdgcn_global_load_async_to_lds_b128(
        (g_vi4p)(xg_e + (size_t)j * 4), (l_vi4p)(&xbuf[0][j * 4]), 0, 0);
  }
#endif

  for (int i = threadIdx.x; i < OUTD * H_DIM; i += blockDim.x)
    wlin_s[i] = W_lin[(size_t)e * OUTD * H_DIM + i];
  if (threadIdx.x < H_DIM) {
    hbuf[threadIdx.x] = 0.0f;
    hh16[threadIdx.x] = (_Float16)0.0f;
  }
  float creg = 0.0f;

  // Preload this wave's 64 rows of W_hh into register A-fragments.
  v16h A[4][8];
  #pragma unroll
  for (int i = 0; i < 4; ++i) {
    int row = (wave * 4 + i) * 16 + l15;
    const float* wr = W_hh + ((size_t)e * G4 + row) * H_DIM;
    #pragma unroll
    for (int kt = 0; kt < 8; ++kt) {
      #pragma unroll
      for (int v = 0; v < 8; ++v) {
        int k = kt * 32 + ((v >> 2) << 4) + (hf << 3) + ((v & 3) << 1);
        A[i][kt][2 * v]     = (_Float16)wr[k];
        A[i][kt][2 * v + 1] = (_Float16)wr[k + 1];
      }
    }
  }
  __syncthreads();

  for (int t = 0; t < B_SEQ; ++t) {
#if USE_ASYNC
    // ---- Async pipeline: issue xg[t+1], ensure xg[t] has landed (in-order,
    //      so ASYNCcnt <= 1 suffices), then publish via the phase-1 barrier.
    if (threadIdx.x >= H_DIM) {
      int j = threadIdx.x - H_DIM;        // 0..255
      if (t + 1 < B_SEQ) {
        __builtin_amdgcn_global_load_async_to_lds_b128(
            (g_vi4p)(xg_e + (size_t)(t + 1) * G4 + j * 4),
            (l_vi4p)(&xbuf[(t + 1) & 1][j * 4]), 0, 0);
      }
      __builtin_amdgcn_s_wait_asynccnt(1);
    }
#else
    const float* xt = xg_e + (size_t)t * G4;
    // Early-issue the gate inputs for this step; WMMA hides the latency.
    float xv0 = 0.f, xv1 = 0.f, xv2 = 0.f, xv3 = 0.f;
    if (threadIdx.x < H_DIM) {
      int j = threadIdx.x;
      xv0 = xt[j];
      xv1 = xt[j + 256];
      xv2 = xt[j + 512];
      xv3 = xt[j + 768];
    } else if (t + 1 < B_SEQ) {
      int j = threadIdx.x - H_DIM;
      __builtin_prefetch(xg_e + (size_t)(t + 1) * G4 + j * 4, 0, 0);
    }
#endif

    // ---- Phase 1: recurrent matvec via WMMA; B built once per K-tile.
    v8f acc0 = {}, acc1 = {}, acc2 = {}, acc3 = {};
    #pragma unroll
    for (int kt = 0; kt < 8; ++kt) {
      v16h b;
      #pragma unroll
      for (int v = 0; v < 8; ++v) {
        int k = kt * 32 + ((v >> 2) << 4) + (hf << 3) + ((v & 3) << 1);
        b[2 * v]     = hh16[k];
        b[2 * v + 1] = hh16[k + 1];
      }
      acc0 = __builtin_amdgcn_wmma_f32_16x16x32_f16(false, A[0][kt], false, b,
                                                    (short)0, acc0, false, false);
      acc1 = __builtin_amdgcn_wmma_f32_16x16x32_f16(false, A[1][kt], false, b,
                                                    (short)0, acc1, false, false);
      acc2 = __builtin_amdgcn_wmma_f32_16x16x32_f16(false, A[2][kt], false, b,
                                                    (short)0, acc2, false, false);
      acc3 = __builtin_amdgcn_wmma_f32_16x16x32_f16(false, A[3][kt], false, b,
                                                    (short)0, acc3, false, false);
    }
    if (l15 == 0) {                  // all N columns identical; lanes 0/16 emit
      int rb = wave * 64 + hf * 8;
      #pragma unroll
      for (int r = 0; r < 8; ++r) gpre[rb + r]      = acc0[r];
      #pragma unroll
      for (int r = 0; r < 8; ++r) gpre[rb + 16 + r] = acc1[r];
      #pragma unroll
      for (int r = 0; r < 8; ++r) gpre[rb + 32 + r] = acc2[r];
      #pragma unroll
      for (int r = 0; r < 8; ++r) gpre[rb + 48 + r] = acc3[r];
    }
    __syncthreads();

    // ---- Phase 2: gate nonlinearity + state update (threads 0..255)
    if (threadIdx.x < H_DIM) {
      int j = threadIdx.x;
#if USE_ASYNC
      const float* xb = xbuf[t & 1];
      float xv0 = xb[j];
      float xv1 = xb[j + 256];
      float xv2 = xb[j + 512];
      float xv3 = xb[j + 768];
#endif
      float gi = gpre[j]       + xv0;
      float gf = gpre[j + 256] + xv1;
      float gg = gpre[j + 512] + xv2;
      float go = gpre[j + 768] + xv3;
      gi = 1.0f / (1.0f + __expf(-gi));
      gf = 1.0f / (1.0f + __expf(-gf));
      go = 1.0f / (1.0f + __expf(-go));
      creg = gf * creg + gi * tanhf(gg);
      float h = go * tanhf(creg);
      hbuf[j] = h;
      hh16[j] = (_Float16)h;
    }
    __syncthreads();

    // ---- Phase 3: fused output projection partials (16 outs x 16 segments)
    if (threadIdx.x < 256) {
      int oo = threadIdx.x & 15, seg = threadIdx.x >> 4;
      float p = 0.0f;
      #pragma unroll
      for (int q = 0; q < 16; ++q)
        p += wlin_s[oo * H_DIM + seg * 16 + q] * hbuf[seg * 16 + q];
      red[oo * 16 + seg] = p;
    }
    __syncthreads();

    // ---- Phase 4: final reduce + gated accumulate into out[t][:]
    if (threadIdx.x < OUTD) {
      float z = b_lin[e * OUTD + threadIdx.x];
      #pragma unroll
      for (int s = 0; s < 16; ++s) z += red[threadIdx.x * 16 + s];
      float gv = gated[t * NE + e];
      if (gv != 0.0f)
        atomicAdd(out + (size_t)t * OUTD + threadIdx.x, gv * z);
    }
    // next-iteration barriers provide the needed ordering for gpre/hh16/red/xbuf
  }
}

// ---------------------------------------------------------------------------
extern "C" void kernel_launch(void* const* d_in, const int* in_sizes, int n_in,
                              void* d_out, int out_size, void* d_ws, size_t ws_size,
                              hipStream_t stream) {
  const float* x     = (const float*)d_in[0];  // [4096,128]
  const float* Wg    = (const float*)d_in[1];  // [16,128]
  const float* bg    = (const float*)d_in[2];  // [16]
  const float* W_ih  = (const float*)d_in[3];  // [16,1024,128]
  const float* W_hh  = (const float*)d_in[4];  // [16,1024,256]
  const float* b_ih  = (const float*)d_in[5];  // [16,1024]
  const float* b_hh  = (const float*)d_in[6];  // [16,1024]
  const float* W_lin = (const float*)d_in[7];  // [16,16,256]
  const float* b_lin = (const float*)d_in[8];  // [16,16]
  float* out = (float*)d_out;                  // [4096,16]

  float* xg    = (float*)d_ws;                            // 16*4096*1024 f32
  float* gated = xg + (size_t)NE * B_SEQ * G4;            // 4096*16 f32

  zero_out_kernel<<<(B_SEQ * OUTD + 255) / 256, 256, 0, stream>>>(out, B_SEQ * OUTD);
  gating_kernel<<<B_SEQ / 256, 256, 0, stream>>>(x, Wg, bg, gated);
  xproj_kernel<<<(NE * 256 * 64) / 8, 256, 0, stream>>>(x, W_ih, b_ih, b_hh, xg);
  lstm_scan_kernel<<<NE, 512, 0, stream>>>(W_hh, W_lin, b_lin, xg, gated, out);
}
